// ALSTMModel_45818711114016
// MI455X (gfx1250) — matrix operand order repaired and checked by reading, not compiled, not run on hardware
//
#include <hip/hip_runtime.h>

#define T_LEN   1024
#define B_SZ    8
#define E_DIM   512
#define H_DIM   512
#define H2_DIM  1024
#define G4      2048
#define NROWS   8192      // B*T
#define NWG_LSTM 16

typedef __attribute__((ext_vector_type(16))) __bf16 v16bf;
typedef __attribute__((ext_vector_type(8)))  float  v8f;

union BfFrag { uint4 u[2]; v16bf v; };

__device__ __forceinline__ unsigned short f2bf(float f) {
  union { float f; unsigned u; } c; c.f = f;
  unsigned r = c.u + 0x7fffu + ((c.u >> 16) & 1u);   // round-to-nearest-even
  return (unsigned short)(r >> 16);
}
__device__ __forceinline__ float bf2f(unsigned short h) {
  union { unsigned u; float f; } c; c.u = ((unsigned)h) << 16;
  return c.f;
}

// Async 16B copy global -> LDS (per-lane), tracked by ASYNCcnt.
__device__ __forceinline__ void async_copy_b128(unsigned lds_off, const void* gaddr) {
  asm volatile("global_load_async_to_lds_b128 %0, %1, off"
               :: "v"(lds_off), "v"(gaddr)
               : "memory");
}
__device__ __forceinline__ void wait_asynccnt0() {
  asm volatile("s_wait_asynccnt 0x0" ::: "memory");
}

// ---------------------------------------------------------------- cvt f32->bf16
__global__ __launch_bounds__(256)
void cvt_f32_to_bf16(const float* __restrict__ s, unsigned short* __restrict__ d, long n) {
  long i = (long)blockIdx.x * 256 + threadIdx.x;
  if (i < n) d[i] = f2bf(s[i]);
}

// ---------------------------------------------------------------- embedding gather
__global__ __launch_bounds__(256)
void embed_gather(const int* __restrict__ idx, const float* __restrict__ emb,
                  unsigned short* __restrict__ x) {
  const int row = blockIdx.x;
  const long id = idx[row];
  const float* src = emb + id * (long)E_DIM;
  unsigned short* dst = x + (long)row * E_DIM;
  for (int c = threadIdx.x; c < E_DIM; c += 256) dst[c] = f2bf(src[c]);
}

// ---------------------------------------------------------------- WMMA GEMM
// C[M,N] = A[M,K](bf16) * W[N,K]^T(bf16) (+bias0+bias1), out f32 or bf16.
// Block tile 128x128, 8 waves of 32x64, K-chunk 32.
// Tiles staged with GLOBAL_LOAD_ASYNC_TO_LDS_B128 (ASYNCcnt), fragments via ds_load_b128.
__global__ __launch_bounds__(256)
void gemm_bf16_wmma(const unsigned short* __restrict__ A, int lda,
                    const unsigned short* __restrict__ W, int ldb,
                    void* __restrict__ Cout, int ldc, int K,
                    const float* __restrict__ bias0,
                    const float* __restrict__ bias1,
                    int out_bf16) {
  __shared__ __align__(16) unsigned short As[128 * 40];
  __shared__ __align__(16) unsigned short Bs[128 * 40];
  const int tid  = threadIdx.x;
  const int wave = tid >> 5;
  const int lane = tid & 31;
  const int m0 = (wave >> 1) * 32;
  const int n0 = (wave & 1) * 64;
  const long rowBase = (long)blockIdx.x * 128;
  const long colBase = (long)blockIdx.y * 128;
  const int tr = tid >> 1;
  const int tc = (tid & 1) * 16;
  const v8f zero8 = {0.f,0.f,0.f,0.f,0.f,0.f,0.f,0.f};
  v8f acc[2][4];
  for (int mi = 0; mi < 2; ++mi)
    for (int ni = 0; ni < 4; ++ni) acc[mi][ni] = zero8;

  const unsigned short* aRow = A + (rowBase + tr) * (long)lda + tc;
  const unsigned short* bRow = W + (colBase + tr) * (long)ldb + tc;
  // workgroup-relative LDS byte offsets for this thread's two 16B staging slots
  const unsigned ldsA0 = (unsigned)(size_t)(As + tr * 40 + tc);
  const unsigned ldsB0 = (unsigned)(size_t)(Bs + tr * 40 + tc);

  for (int kk = 0; kk < K; kk += 32) {
    __syncthreads();                            // previous reads of the tile done
    async_copy_b128(ldsA0,      aRow + kk);
    async_copy_b128(ldsA0 + 16, aRow + kk + 8);
    async_copy_b128(ldsB0,      bRow + kk);
    async_copy_b128(ldsB0 + 16, bRow + kk + 8);
    if (kk + 32 < K) {                          // global_prefetch_b8 of next slab
      __builtin_prefetch(aRow + kk + 32, 0, 1);
      __builtin_prefetch(bRow + kk + 32, 0, 1);
    }
    wait_asynccnt0();                           // this wave's async copies landed
    __syncthreads();                            // all waves' copies landed

    const int fm = lane & 15;
    const int kh = lane >> 4;
    BfFrag af[2], bfv[4];
    #pragma unroll
    for (int mi = 0; mi < 2; ++mi) {            // A frag: lanes0-15 K 0..7/16..23
      int r = m0 + mi * 16 + fm;
      af[mi].u[0] = *(const uint4*)(As + r * 40 + kh * 8);
      af[mi].u[1] = *(const uint4*)(As + r * 40 + kh * 8 + 16);
    }
    #pragma unroll
    for (int ni = 0; ni < 4; ++ni) {            // B frag: lanes0-15 K 0..7/16..23
      int r = n0 + ni * 16 + fm;
      bfv[ni].u[0] = *(const uint4*)(Bs + r * 40 + kh * 8);
      bfv[ni].u[1] = *(const uint4*)(Bs + r * 40 + kh * 8 + 16);
    }
    #pragma unroll
    for (int mi = 0; mi < 2; ++mi)
      #pragma unroll
      for (int ni = 0; ni < 4; ++ni)
        acc[mi][ni] = __builtin_amdgcn_wmma_f32_16x16x32_bf16(
            false, af[mi].v, false, bfv[ni].v, (short)0, acc[mi][ni], false, false);
  }

  const int cn = lane & 15;
  const int mh = (lane >> 4) * 8;
  for (int ni = 0; ni < 4; ++ni) {
    long col = colBase + n0 + ni * 16 + cn;
    float badd = 0.f;
    if (bias0) badd += bias0[col];
    if (bias1) badd += bias1[col];
    for (int mi = 0; mi < 2; ++mi) {
      long row0 = rowBase + m0 + mi * 16 + mh;
      #pragma unroll
      for (int r = 0; r < 8; ++r) {
        float v = acc[mi][ni][r] + badd;
        long idx = (row0 + r) * (long)ldc + col;
        if (out_bf16) ((unsigned short*)Cout)[idx] = f2bf(v);
        else          ((float*)Cout)[idx] = v;
      }
    }
  }
}

// ---------------------------------------------------------------- persistent LSTM direction
// 16 WGs; WG owns 32 hidden cols (128 gate rows). Whh rows live in VGPRs.
// h double-buffered [2][16][512] bf16, rows 8..15 stay zero (EXEC all-1 for WMMA).
__global__ __launch_bounds__(256)
void lstm_dir_wmma(const float* __restrict__ xw,            // [B*T, 2048] preact (x*Wih^T + biases)
                   const unsigned short* __restrict__ Whh,  // [2048, 512] bf16, this direction
                   unsigned short* __restrict__ hstate,     // [2][16][512] bf16
                   unsigned short* __restrict__ out,        // bf16 base + dir*512, row stride 1024
                   int reverse,
                   unsigned* __restrict__ bar) {
  __shared__ float gpost[4][8][32];
  const int tid  = threadIdx.x;
  const int lane = tid & 31;
  const int wave = tid >> 5;
  const int j0 = (int)blockIdx.x * 32;
  const int fm = lane & 15;
  const int kh = lane >> 4;
  const int g  = wave >> 1;                       // gate handled by this wave
  const int jl = (wave * 16 + fm) & 31;           // local hidden col of this lane's gate col
  const long ncol = (long)g * 512 + j0 + jl;      // gate row in Whh == gate col in xw

  // Preload the 16 Whh B-fragments (this wave's 16 gate rows) into registers.
  BfFrag breg[16];
  #pragma unroll
  for (int ki = 0; ki < 16; ++ki) {
    const unsigned short* wr = Whh + ncol * 512 + ki * 32 + kh * 8;
    breg[ki].u[0] = *(const uint4*)(wr);
    breg[ki].u[1] = *(const uint4*)(wr + 16);
  }

  const int cb = wave;        // batch index for c/h update (8 waves)
  const int cj = lane;        // hidden col within slice (32 lanes)
  float cstate = 0.f;
  const v8f zero8 = {0.f,0.f,0.f,0.f,0.f,0.f,0.f,0.f};

  for (int s = 0; s < T_LEN; ++s) {
    const int t = reverse ? (T_LEN - 1 - s) : s;
    const unsigned short* hread  = hstate + (s & 1) * (16 * 512);
    unsigned short*       hwrite = hstate + ((s & 1) ^ 1) * (16 * 512);

    v8f acc = zero8;
    #pragma unroll
    for (int ki = 0; ki < 16; ++ki) {
      BfFrag af;
      const unsigned short* hr = hread + fm * 512 + ki * 32 + kh * 8;
      af.u[0] = *(const uint4*)(hr);
      af.u[1] = *(const uint4*)(hr + 16);
      acc = __builtin_amdgcn_wmma_f32_16x16x32_bf16(
          false, af.v, false, breg[ki].v, (short)0, acc, false, false);
    }

    // add x-part preactivation, apply gate nonlinearity, stash to LDS
    #pragma unroll
    for (int r = 0; r < 8; ++r) {
      int m = kh * 8 + r;                 // C/D row = batch index
      if (m < 8) {
        float p = acc[r] + xw[((long)m * T_LEN + t) * (long)G4 + ncol];
        float a = (g == 2) ? tanhf(p) : (1.f / (1.f + __expf(-p)));
        gpost[g][m][jl] = a;
      }
    }
    __syncthreads();
    {
      float iv = gpost[0][cb][cj];
      float fv = gpost[1][cb][cj];
      float gv = gpost[2][cb][cj];
      float ov = gpost[3][cb][cj];
      cstate = fv * cstate + iv * gv;
      float h = ov * tanhf(cstate);
      unsigned short hb = f2bf(h);
      hwrite[cb * 512 + j0 + cj] = hb;
      out[((long)cb * T_LEN + t) * (long)H2_DIM + j0 + cj] = hb;
    }
    __threadfence();
    __syncthreads();
    if (tid == 0) {                       // device-scope split barrier across 16 WGs
      __hip_atomic_fetch_add(bar, 1u, __ATOMIC_RELEASE, __HIP_MEMORY_SCOPE_AGENT);
      const unsigned tgt = (unsigned)NWG_LSTM * (unsigned)(s + 1);
      while (__hip_atomic_load(bar, __ATOMIC_ACQUIRE, __HIP_MEMORY_SCOPE_AGENT) < tgt)
        __builtin_amdgcn_s_sleep(1);
    }
    __syncthreads();
  }
}

// ---------------------------------------------------------------- last-query attention
// block = (half, b, head). q from last row; softmax over 1024 keys; ctx -> [2][8][512] f32
__global__ __launch_bounds__(256)
void attn_last(const unsigned short* __restrict__ xcat,   // [8192][1024] bf16
               const unsigned short* __restrict__ kbf,    // [2][8192][512] bf16
               const unsigned short* __restrict__ vbf,    // [2][8192][512] bf16
               const float* __restrict__ Wq_f, const float* __restrict__ Wq_b,
               float* __restrict__ ctx) {
  __shared__ float q[64];
  __shared__ float sc[1024];
  __shared__ float red[256];
  const int tid = threadIdx.x;
  const int half = blockIdx.x >> 6;
  const int rem  = blockIdx.x & 63;
  const int b = rem >> 3;
  const int h = rem & 7;
  const float* Wq = half ? Wq_b : Wq_f;
  const unsigned short* xrow = xcat + ((long)(b * T_LEN + (T_LEN - 1))) * H2_DIM + half * H_DIM;

  if (tid < 64) {
    float a = 0.f;
    for (int k = 0; k < H_DIM; ++k) a += bf2f(xrow[k]) * Wq[(h * 64 + tid) * H_DIM + k];
    q[tid] = a;
  }
  __syncthreads();

  const unsigned short* kb = kbf + (long)half * NROWS * H_DIM;
  float myv[4];
  float lmax = -1e30f;
  for (int j = 0; j < 4; ++j) {
    int tt = tid * 4 + j;
    const unsigned short* kr = kb + ((long)(b * T_LEN + tt)) * H_DIM + h * 64;
    float a = 0.f;
    for (int d = 0; d < 64; ++d) a += q[d] * bf2f(kr[d]);
    a *= 0.125f;                        // 1/sqrt(64)
    myv[j] = a;
    lmax = fmaxf(lmax, a);
  }
  red[tid] = lmax; __syncthreads();
  for (int o = 128; o; o >>= 1) { if (tid < o) red[tid] = fmaxf(red[tid], red[tid + o]); __syncthreads(); }
  const float mx = red[0];
  __syncthreads();
  float ls = 0.f;
  for (int j = 0; j < 4; ++j) { float e = __expf(myv[j] - mx); sc[tid * 4 + j] = e; ls += e; }
  red[tid] = ls; __syncthreads();
  for (int o = 128; o; o >>= 1) { if (tid < o) red[tid] += red[tid + o]; __syncthreads(); }
  const float inv = 1.f / red[0];
  __syncthreads();

  if (tid < 64) {
    const unsigned short* vb = vbf + (long)half * NROWS * H_DIM;
    float a = 0.f;
    for (int tt = 0; tt < T_LEN; ++tt)
      a += sc[tt] * bf2f(vb[((long)(b * T_LEN + tt)) * H_DIM + h * 64 + tid]);
    ctx[((long)(half * 8 + b)) * H_DIM + h * 64 + tid] = a * inv;
  }
}

// ---------------------------------------------------------------- final head (8 rows only)
__global__ __launch_bounds__(256)
void final_head(const float* __restrict__ ctx,
                const float* __restrict__ Wo_f, const float* __restrict__ bo_f,
                const float* __restrict__ Wo_b, const float* __restrict__ bo_b,
                const float* __restrict__ lng, const float* __restrict__ lnb,
                const float* __restrict__ W1, const float* __restrict__ b1,
                const float* __restrict__ W2, const float* __restrict__ b2,
                const float* __restrict__ Wout, const float* __restrict__ bout,
                float* __restrict__ out) {
  __shared__ float xc[1024];
  __shared__ float a1[2048];
  __shared__ float y2[1024];
  __shared__ float red[256];
  const int b = blockIdx.x, tid = threadIdx.x;

  for (int c = tid; c < 1024; c += 256) {       // attention out-proj (concat halves)
    int half = c >> 9, cc = c & 511;
    const float* Wo = half ? Wo_b : Wo_f;
    const float* bo = half ? bo_b : bo_f;
    const float* cx = ctx + ((long)(half * 8 + b)) * 512;
    float a = bo[cc];
    for (int k = 0; k < 512; ++k) a += cx[k] * Wo[(long)cc * 512 + k];
    xc[c] = a;
  }
  __syncthreads();
  float s = 0.f, s2 = 0.f;
  for (int c = tid; c < 1024; c += 256) { float v = xc[c]; s += v; s2 += v * v; }
  red[tid] = s; __syncthreads();
  for (int o = 128; o; o >>= 1) { if (tid < o) red[tid] += red[tid + o]; __syncthreads(); }
  const float mu = red[0] * (1.f / 1024.f);
  __syncthreads();
  red[tid] = s2; __syncthreads();
  for (int o = 128; o; o >>= 1) { if (tid < o) red[tid] += red[tid + o]; __syncthreads(); }
  const float var = red[0] * (1.f / 1024.f) - mu * mu;
  const float rstd = rsqrtf(var + 1e-5f);
  __syncthreads();
  for (int c = tid; c < 1024; c += 256) xc[c] = (xc[c] - mu) * rstd * lng[c] + lnb[c];
  __syncthreads();
  for (int o = tid; o < 2048; o += 256) {       // FF1 + relu
    float a = b1[o];
    for (int k = 0; k < 1024; ++k) a += xc[k] * W1[(long)o * 1024 + k];
    a1[o] = fmaxf(a, 0.f);
  }
  __syncthreads();
  for (int o = tid; o < 1024; o += 256) {       // FF2
    float a = b2[o];
    for (int k = 0; k < 2048; ++k) a += a1[k] * W2[(long)o * 2048 + k];
    y2[o] = a;
  }
  __syncthreads();
  if (tid < 6) {                                // logits
    float a = bout[tid];
    for (int k = 0; k < 1024; ++k) a += y2[k] * Wout[(long)tid * 1024 + k];
    out[b * 6 + tid] = a;
  }
}

// ---------------------------------------------------------------- launcher
extern "C" void kernel_launch(void* const* d_in, const int* in_sizes, int n_in,
                              void* d_out, int out_size, void* d_ws, size_t ws_size,
                              hipStream_t stream) {
  (void)in_sizes; (void)n_in; (void)out_size; (void)ws_size;
  const int*   in_idx = (const int*)  d_in[0];
  const float* emb    = (const float*)d_in[1];
  const float* Wih0   = (const float*)d_in[2];
  const float* Whh0   = (const float*)d_in[3];
  const float* bih0   = (const float*)d_in[4];
  const float* bhh0   = (const float*)d_in[5];
  const float* Wih1   = (const float*)d_in[6];
  const float* Whh1   = (const float*)d_in[7];
  const float* bih1   = (const float*)d_in[8];
  const float* bhh1   = (const float*)d_in[9];
  const float* af_Wq  = (const float*)d_in[10];
  const float* af_Wk  = (const float*)d_in[11];
  const float* af_Wv  = (const float*)d_in[12];
  const float* af_Wo  = (const float*)d_in[13];
  const float* af_bo  = (const float*)d_in[14];
  const float* ab_Wq  = (const float*)d_in[15];
  const float* ab_Wk  = (const float*)d_in[16];
  const float* ab_Wv  = (const float*)d_in[17];
  const float* ab_Wo  = (const float*)d_in[18];
  const float* ab_bo  = (const float*)d_in[19];
  const float* lng    = (const float*)d_in[20];
  const float* lnb    = (const float*)d_in[21];
  const float* ffW1   = (const float*)d_in[22];
  const float* ffb1   = (const float*)d_in[23];
  const float* ffW2   = (const float*)d_in[24];
  const float* ffb2   = (const float*)d_in[25];
  const float* outW   = (const float*)d_in[26];
  const float* outb   = (const float*)d_in[27];

  char* ws = (char*)d_ws;
  size_t off = 0;
  auto take = [&](size_t bytes) -> void* {
    void* p = ws + off;
    off = (off + bytes + 255) & ~(size_t)255;
    return p;
  };
  unsigned short* xemb  = (unsigned short*)take((size_t)NROWS * E_DIM * 2);
  unsigned short* wih0b = (unsigned short*)take((size_t)2 * G4 * E_DIM * 2);
  unsigned short* whh0b = (unsigned short*)take((size_t)2 * G4 * H_DIM * 2);
  unsigned short* wih1b = (unsigned short*)take((size_t)2 * G4 * H2_DIM * 2);
  unsigned short* whh1b = (unsigned short*)take((size_t)2 * G4 * H_DIM * 2);
  unsigned short* kwbf  = (unsigned short*)take((size_t)2 * H_DIM * H_DIM * 2);
  unsigned short* vwbf  = (unsigned short*)take((size_t)2 * H_DIM * H_DIM * 2);
  float*          xw    = (float*)take((size_t)NROWS * G4 * 4);
  unsigned short* h0cat = (unsigned short*)take((size_t)NROWS * H2_DIM * 2);
  unsigned short* h1cat = (unsigned short*)take((size_t)NROWS * H2_DIM * 2);
  unsigned short* kbf   = (unsigned short*)take((size_t)2 * NROWS * H_DIM * 2);
  unsigned short* vbf   = (unsigned short*)take((size_t)2 * NROWS * H_DIM * 2);
  unsigned short* hst   = (unsigned short*)take((size_t)2 * 16 * 512 * 2);   // 32768 B
  unsigned*       bar   = (unsigned*)take(256);                              // adjacent to hst
  float*          ctx   = (float*)take((size_t)2 * 8 * 512 * 4);

  auto cvt = [&](const float* s, unsigned short* d, long n) {
    cvt_f32_to_bf16<<<dim3((unsigned)((n + 255) / 256)), dim3(256), 0, stream>>>(s, d, n);
  };
  cvt(Wih0, wih0b, 2L * G4 * E_DIM);
  cvt(Whh0, whh0b, 2L * G4 * H_DIM);
  cvt(Wih1, wih1b, 2L * G4 * H2_DIM);
  cvt(Whh1, whh1b, 2L * G4 * H_DIM);
  cvt(af_Wk, kwbf,                 (long)H_DIM * H_DIM);
  cvt(ab_Wk, kwbf + H_DIM * H_DIM, (long)H_DIM * H_DIM);
  cvt(af_Wv, vwbf,                 (long)H_DIM * H_DIM);
  cvt(ab_Wv, vwbf + H_DIM * H_DIM, (long)H_DIM * H_DIM);

  embed_gather<<<dim3(NROWS), dim3(256), 0, stream>>>(in_idx, emb, xemb);

  for (int layer = 0; layer < 2; ++layer) {
    for (int dir = 0; dir < 2; ++dir) {
      const unsigned short* Ap  = layer ? h0cat : xemb;
      const int K = layer ? H2_DIM : E_DIM;
      const unsigned short* Wp  = (layer ? wih1b : wih0b) + (size_t)dir * G4 * K;
      const float* bA = (layer ? bih1 : bih0) + dir * G4;
      const float* bB = (layer ? bhh1 : bhh0) + dir * G4;
      gemm_bf16_wmma<<<dim3(NROWS / 128, G4 / 128), dim3(256), 0, stream>>>(
          Ap, K, Wp, K, xw, G4, K, bA, bB, 0);
      hipMemsetAsync(hst, 0, 32768 + 256, stream);   // h double-buffer + barrier ctr
      const unsigned short* Whp = (layer ? whh1b : whh0b) + (size_t)dir * G4 * H_DIM;
      unsigned short* outp = (layer ? h1cat : h0cat) + dir * H_DIM;
      lstm_dir_wmma<<<dim3(NWG_LSTM), dim3(256), 0, stream>>>(xw, Whp, hst, outp, dir, bar);
    }
  }

  for (int half = 0; half < 2; ++half) {
    const unsigned short* Ap = h1cat + half * H_DIM;
    gemm_bf16_wmma<<<dim3(NROWS / 128, H_DIM / 128), dim3(256), 0, stream>>>(
        Ap, H2_DIM, kwbf + (size_t)half * H_DIM * H_DIM, H_DIM,
        kbf + (size_t)half * NROWS * H_DIM, H_DIM, H_DIM, nullptr, nullptr, 1);
    gemm_bf16_wmma<<<dim3(NROWS / 128, H_DIM / 128), dim3(256), 0, stream>>>(
        Ap, H2_DIM, vwbf + (size_t)half * H_DIM * H_DIM, H_DIM,
        vbf + (size_t)half * NROWS * H_DIM, H_DIM, H_DIM, nullptr, nullptr, 1);
  }

  attn_last<<<dim3(128), dim3(256), 0, stream>>>(h1cat, kbf, vbf, af_Wq, ab_Wq, ctx);

  final_head<<<dim3(B_SZ), dim3(256), 0, stream>>>(
      ctx, af_Wo, af_bo, ab_Wo, ab_bo, lng, lnb, ffW1, ffb1, ffW2, ffb2, outW, outb,
      (float*)d_out);
}
